// LIIFNet_53429393162697
// MI455X (gfx1250) — matrix operand order repaired and checked by reading, not compiled, not run on hardware
//
#include <hip/hip_runtime.h>

typedef _Float16 half_t;
typedef _Float16 v16h __attribute__((ext_vector_type(16)));
typedef _Float16 v8h  __attribute__((ext_vector_type(8)));
typedef float    v8f  __attribute__((ext_vector_type(8)));

// Problem constants (from reference)
constexpr int BB   = 4;
constexpr int CC   = 64;
constexpr int HH   = 64;
constexpr int WW   = 64;
constexpr int QQ   = 192 * 192;
constexpr int NTOT = BB * QQ;        // 147456 rows per corner
constexpr int IND  = 580;
constexpr int K0P  = 608;            // IN_DIM padded to 19*32
constexpr int HID  = 256;
constexpr int TM   = 64;             // rows per block
constexpr int HS   = 264;            // LDS activation row stride (halfs)

constexpr int KT0 = K0P / 32;        // 19 k-tiles, layer 0
constexpr int KTH = HID / 32;        // 8 k-tiles, hidden layers

// Packed-weight fragment sizes (halfs): frag = 32 lanes x 16 halfs = 512
constexpr int W0P_HALFS = 16 * KT0 * 512;   // 155648
constexpr int WHP_HALFS = 16 * KTH * 512;   // 65536
constexpr int W4P_HALFS = 1 * KTH * 512;    // 4096

// LDS layout (bytes)
constexpr int OFF_SCAL = 0;                        // scalars: 1536 B
constexpr int OFF_X    = 1536;                     // 64 x 608 halfs = 77824 B
constexpr int OFF_H0   = OFF_X + TM * K0P * 2;     // 64 x 264 halfs = 33792 B
constexpr int OFF_H1   = OFF_H0 + TM * HS * 2;
constexpr int SMEM_BYTES = OFF_H1 + TM * HS * 2;   // 146944 B (2 blocks / 320KB WGP)

#define WMMA_F16(a, b, c) \
    __builtin_amdgcn_wmma_f32_16x16x32_f16(false, (a), false, (b), (short)0, (c), false, false)

// ---------------- weight prep ----------------
// Pack weights into this kernel's WMMA-B fragment order: fragment f = ntile*KT + ktile
// holds, contiguously, lane 0..31 x 16 halfs.  B k order used here (contiguous): lanes 0-15 hold
// col N=lane, K 0..15 contiguous; lanes 16-31 hold col N=lane-16, K 16..31.
// => each B-frag load is one wave-contiguous 1KB burst.
__global__ void liif_prep_weights(const float* __restrict__ w0,
                                  const float* __restrict__ w1,
                                  const float* __restrict__ w2,
                                  const float* __restrict__ w3,
                                  const float* __restrict__ w4,
                                  half_t* __restrict__ W0p,
                                  half_t* __restrict__ W1p,
                                  half_t* __restrict__ W2p,
                                  half_t* __restrict__ W3p,
                                  half_t* __restrict__ W4p) {
    int idx = blockIdx.x * blockDim.x + threadIdx.x;
    // region 0: layer 0, (580->608) x 256, src w0 is (580 x 256)
    if (idx < W0P_HALFS) {
        const int f = idx >> 9, w = idx & 511;
        const int lane = w >> 4, j = w & 15;
        const int nt = f / KT0, kt = f - nt * KT0;
        const int ncol = nt * 16 + (lane & 15);
        const int k = kt * 32 + ((lane >> 4) << 4) + j;
        W0p[idx] = (half_t)(k < IND ? w0[k * HID + ncol] : 0.0f);
        return;
    }
    idx -= W0P_HALFS;
    // region 1: layers 1..3, 256 x 256
    if (idx < 3 * WHP_HALFS) {
        const int L = idx / WHP_HALFS;
        const int r2 = idx - L * WHP_HALFS;
        const int f = r2 >> 9, w = r2 & 511;
        const int lane = w >> 4, j = w & 15;
        const int nt = f >> 3, kt = f & 7;
        const int ncol = nt * 16 + (lane & 15);
        const int k = kt * 32 + ((lane >> 4) << 4) + j;
        const float* src = (L == 0) ? w1 : (L == 1) ? w2 : w3;
        half_t* dst = (L == 0) ? W1p : (L == 1) ? W2p : W3p;
        dst[r2] = (half_t)src[k * HID + ncol];
        return;
    }
    idx -= 3 * WHP_HALFS;
    // region 2: layer 4, 256 x (3->16), src w4 is (256 x 3)
    if (idx < W4P_HALFS) {
        const int f = idx >> 9, w = idx & 511;     // f = ktile
        const int lane = w >> 4, j = w & 15;
        const int ncol = lane & 15;
        const int k = f * 32 + ((lane >> 4) << 4) + j;
        W4p[idx] = (half_t)(ncol < 3 ? w4[k * 3 + ncol] : 0.0f);
    }
}

// ---------------- WMMA fragment loaders ----------------
// A operand (16x32), k order used here: lanes 0-15 rows M=0..15 hold K 0..15 contiguous;
// lanes 16-31 same rows hold K 16..31 (matches packed B).
__device__ __forceinline__ v16h load_a_frag(const half_t* base, int stride, int m, int k0, int lane) {
    const half_t* p = base + (size_t)m * stride + k0 + ((lane >> 4) << 4);
    v8h lo = *(const v8h*)p;
    v8h hi = *(const v8h*)(p + 8);
    return __builtin_shufflevector(lo, hi, 0, 1, 2, 3, 4, 5, 6, 7, 8, 9, 10, 11, 12, 13, 14, 15);
}

// Packed B fragment: wave-contiguous 1KB per (ntile, ktile)
__device__ __forceinline__ v16h load_b_packed(const half_t* Wp, int frag, int lane) {
    return *(const v16h*)(Wp + (((size_t)frag << 5) + lane) * 16);
}

// One GEMM+bias+ReLU layer: (64 x K) @ (K x 256) -> dst LDS (64 x 256)
// No internal barrier; caller syncs after.
__device__ __forceinline__ void gemm_relu_layer(const half_t* __restrict__ src, int sstride,
                                                const half_t* __restrict__ Wp, int KT,
                                                const float* __restrict__ bias,
                                                half_t* __restrict__ dst,
                                                int wrow, int wcol, int lane) {
    v8f acc[8] = {};
    const int mrow = wrow * 16 + (lane & 15);
    for (int kt = 0; kt < KT; ++kt) {
        const v16h a = load_a_frag(src, sstride, mrow, kt * 32, lane);
#pragma unroll
        for (int nt = 0; nt < 8; ++nt) {
            const v16h bf = load_b_packed(Wp, (wcol * 8 + nt) * KT + kt, lane);
            acc[nt] = WMMA_F16(a, bf, acc[nt]);
        }
    }
#pragma unroll
    for (int nt = 0; nt < 8; ++nt) {
        const int ncol = wcol * 128 + nt * 16 + (lane & 15);
        const float b = bias[ncol];
#pragma unroll
        for (int r = 0; r < 8; ++r) {
            const int mm = wrow * 16 + ((lane >> 4) << 3) + r;
            dst[mm * HS + ncol] = (half_t)fmaxf(acc[nt][r] + b, 0.0f);
        }
    }
}

// ---------------- fused gather + 5-layer MLP ----------------
__global__ __launch_bounds__(256)
void liif_fused(const float* __restrict__ feat,
                const float* __restrict__ coord,
                const float* __restrict__ cell,
                const half_t* __restrict__ W0p, const float* __restrict__ b0,
                const half_t* __restrict__ W1p, const float* __restrict__ b1,
                const half_t* __restrict__ W2p, const float* __restrict__ b2,
                const half_t* __restrict__ W3p, const float* __restrict__ b3,
                const half_t* __restrict__ W4p, const float* __restrict__ b4,
                float* __restrict__ preds, float* __restrict__ areas) {
    extern __shared__ char smem[];
    float* s_relx = (float*)(smem + OFF_SCAL);
    float* s_rely = s_relx + TM;
    float* s_clx  = s_rely + TM;
    float* s_cly  = s_clx + TM;
    int*   s_iy   = (int*)(s_cly + TM);
    int*   s_ix   = s_iy + TM;
    half_t* sX  = (half_t*)(smem + OFF_X);
    half_t* sH0 = (half_t*)(smem + OFF_H0);
    half_t* sH1 = (half_t*)(smem + OFF_H1);

    const int corner  = blockIdx.y;
    const int rowBase = blockIdx.x * TM;
    const int tid  = threadIdx.x;
    const int lane = tid & 31;
    const int wid  = tid >> 5;
    const int wrow = wid & 3;   // 4 row-waves x 16 rows
    const int wcol = wid >> 2;  // 2 col-waves x 128 cols

    // ---- Stage A: per-row sampling scalars + area term ----
    if (tid < TM) {
        const int r = tid;
        const int n = rowBase + r;
        const float cox = coord[(size_t)n * 2 + 0];
        const float coy = coord[(size_t)n * 2 + 1];
        const float clx = cell[(size_t)n * 2 + 0] * (float)HH;
        const float cly = cell[(size_t)n * 2 + 1] * (float)WW;
        const float vx = (corner >> 1) ? 1.0f : -1.0f;
        const float vy = (corner & 1) ? 1.0f : -1.0f;
        const float rx = 1.0f / (float)HH, ry = 1.0f / (float)WW;
        const float E = 1e-6f;
        float cx = fminf(fmaxf(cox + vx * rx + E, -1.0f + E), 1.0f - E);
        float cy = fminf(fmaxf(coy + vy * ry + E, -1.0f + E), 1.0f - E);
        int iy = min(max((int)floorf(((cx + 1.0f) * (float)HH - 1.0f) * 0.5f + 0.5f), 0), HH - 1);
        int ix = min(max((int)floorf(((cy + 1.0f) * (float)WW - 1.0f) * 0.5f + 0.5f), 0), WW - 1);
        const float qcx = ((float)iy + 0.5f) * (2.0f / (float)HH) - 1.0f;
        const float qcy = ((float)ix + 0.5f) * (2.0f / (float)WW) - 1.0f;
        const float relx = (cox - qcx) * (float)HH;
        const float rely = (coy - qcy) * (float)WW;
        s_relx[r] = relx; s_rely[r] = rely;
        s_clx[r] = clx;   s_cly[r] = cly;
        s_iy[r] = iy;     s_ix[r] = ix;
        areas[(size_t)corner * NTOT + n] = fabsf(relx * rely) + 1e-9f;
    }
    __syncthreads();

    // ---- Stage B: gather 64 x 608 f16 input tile into LDS ----
    for (int idx = tid; idx < TM * K0P; idx += 256) {
        const int r = idx / K0P;
        const int k = idx - r * K0P;
        float v = 0.0f;
        if (k < CC * 9) {
            const int ch = k / 9;
            const int t  = k - ch * 9;
            const int kh = t / 3;
            const int kw = t - kh * 3;
            const int y = s_iy[r] + kh - 1;
            const int x = s_ix[r] + kw - 1;
            if ((unsigned)y < (unsigned)HH && (unsigned)x < (unsigned)WW) {
                const int b = (rowBase + r) / QQ;
                v = feat[(((size_t)b * CC + ch) * HH + y) * WW + x];
            }
        } else if (k == 576) v = s_relx[r];
        else if (k == 577) v = s_rely[r];
        else if (k == 578) v = s_clx[r];
        else if (k == 579) v = s_cly[r];
        sX[r * K0P + k] = (half_t)v;
    }
    __syncthreads();

    // ---- 4 hidden layers, ping-pong activations (one barrier per layer) ----
    gemm_relu_layer(sX,  K0P, W0p, KT0, b0, sH0, wrow, wcol, lane);
    __syncthreads();
    gemm_relu_layer(sH0, HS,  W1p, KTH, b1, sH1, wrow, wcol, lane);
    __syncthreads();
    gemm_relu_layer(sH1, HS,  W2p, KTH, b2, sH0, wrow, wcol, lane);
    __syncthreads();
    gemm_relu_layer(sH0, HS,  W3p, KTH, b3, sH1, wrow, wcol, lane);
    __syncthreads();

    // ---- Layer 4: (64 x 256) @ (256 x 16-padded), only col-wave 0 ----
    if (wcol == 0) {
        v8f acc = {};
        const int mrow = wrow * 16 + (lane & 15);
        for (int kt = 0; kt < KTH; ++kt) {
            const v16h a  = load_a_frag(sH1, HS, mrow, kt * 32, lane);
            const v16h bf = load_b_packed(W4p, kt, lane);
            acc = WMMA_F16(a, bf, acc);
        }
        const int n = lane & 15;
        if (n < 3) {
            const float bias = b4[n];
#pragma unroll
            for (int r = 0; r < 8; ++r) {
                const int mm  = wrow * 16 + ((lane >> 4) << 3) + r;
                const int row = rowBase + mm;
                preds[((size_t)corner * NTOT + row) * 3 + n] = acc[r] + bias;
            }
        }
    }
}

// ---------------- 4-corner area-weighted combine ----------------
__global__ void liif_combine(const float* __restrict__ preds,
                             const float* __restrict__ areas,
                             float* __restrict__ out) {
    const int n = blockIdx.x * blockDim.x + threadIdx.x;
    if (n >= NTOT) return;
    float a0 = areas[0 * (size_t)NTOT + n];
    float a1 = areas[1 * (size_t)NTOT + n];
    float a2 = areas[2 * (size_t)NTOT + n];
    float a3 = areas[3 * (size_t)NTOT + n];
    const float inv = 1.0f / (a0 + a1 + a2 + a3);
    // local_ensemble swaps: weight for pred i is areas[order[i]], order = (3,2,1,0)
    const float w0 = a3 * inv, w1 = a2 * inv, w2 = a1 * inv, w3 = a0 * inv;
#pragma unroll
    for (int j = 0; j < 3; ++j) {
        float s = preds[(0 * (size_t)NTOT + n) * 3 + j] * w0
                + preds[(1 * (size_t)NTOT + n) * 3 + j] * w1
                + preds[(2 * (size_t)NTOT + n) * 3 + j] * w2
                + preds[(3 * (size_t)NTOT + n) * 3 + j] * w3;
        out[(size_t)n * 3 + j] = s;
    }
}

extern "C" void kernel_launch(void* const* d_in, const int* in_sizes, int n_in,
                              void* d_out, int out_size, void* d_ws, size_t ws_size,
                              hipStream_t stream) {
    const float* feat  = (const float*)d_in[0];
    const float* coord = (const float*)d_in[1];
    const float* cell  = (const float*)d_in[2];
    const float* w0 = (const float*)d_in[3];
    const float* b0 = (const float*)d_in[4];
    const float* w1 = (const float*)d_in[5];
    const float* b1 = (const float*)d_in[6];
    const float* w2 = (const float*)d_in[7];
    const float* b2 = (const float*)d_in[8];
    const float* w3 = (const float*)d_in[9];
    const float* b3 = (const float*)d_in[10];
    const float* w4 = (const float*)d_in[11];
    const float* b4 = (const float*)d_in[12];
    float* out = (float*)d_out;

    // workspace layout
    char* ws = (char*)d_ws;
    size_t off = 0;
    half_t* W0p = (half_t*)(ws + off); off += (size_t)W0P_HALFS * 2;   // 311296 B
    half_t* W1p = (half_t*)(ws + off); off += (size_t)WHP_HALFS * 2;   // 131072 B
    half_t* W2p = (half_t*)(ws + off); off += (size_t)WHP_HALFS * 2;
    half_t* W3p = (half_t*)(ws + off); off += (size_t)WHP_HALFS * 2;
    half_t* W4p = (half_t*)(ws + off); off += (size_t)W4P_HALFS * 2;   // 8192 B
    float* preds = (float*)(ws + off); off += (size_t)4 * NTOT * 3 * 4;
    float* areas = (float*)(ws + off); off += (size_t)4 * NTOT * 4;
    (void)ws_size; (void)n_in; (void)in_sizes; (void)out_size;

    // 1) weight conversion into packed WMMA-B fragment order
    {
        const int total = W0P_HALFS + 3 * WHP_HALFS + W4P_HALFS;
        liif_prep_weights<<<(total + 255) / 256, 256, 0, stream>>>(
            w0, w1, w2, w3, w4, W0p, W1p, W2p, W3p, W4p);
    }

    // 2) fused gather + MLP (dynamic LDS > 64KB: raise the cap)
    hipFuncSetAttribute((const void*)liif_fused,
                        hipFuncAttributeMaxDynamicSharedMemorySize, SMEM_BYTES);
    {
        dim3 grid(NTOT / TM, 4);
        liif_fused<<<grid, 256, SMEM_BYTES, stream>>>(
            feat, coord, cell,
            W0p, b0, W1p, b1, W2p, b2, W3p, b3, W4p, b4,
            preds, areas);
    }

    // 3) ensemble combine
    liif_combine<<<(NTOT + 255) / 256, 256, 0, stream>>>(preds, areas, out);
}
